// FactorizedAttention_70961449665079
// MI455X (gfx1250) — matrix operand order repaired and checked by reading, not compiled, not run on hardware
//
#include <hip/hip_runtime.h>

typedef __attribute__((ext_vector_type(16))) _Float16 v16h;
typedef __attribute__((ext_vector_type(8)))  _Float16 v8h;
typedef __attribute__((ext_vector_type(4)))  _Float16 v4h;
typedef __attribute__((ext_vector_type(8)))  float    v8f;
typedef __attribute__((ext_vector_type(4)))  float    f4;

#define BB    4
#define CC    128
#define DD    64
#define NTOK  1024          // h*w = 32*32
#define HEADS 8
#define DH    32
#define HID   256           // HEADS*DH
#define QKVE  768           // 3*HID

union AFrag { v16h v; v8h h[2]; };

// A-matrix tile 16x32 f16 (M=rows, K=cols) from row-major src (ld in halves).
// ISA layout: lanes 0-15: M=lane, K {k0..k0+7, k0+16..k0+23};
//             lanes16-31: M=lane-16, K {k0+8..k0+15, k0+24..k0+31}.
__device__ __forceinline__ v16h load_tileA(const _Float16* src, int ld, int m0, int k0, int lane) {
    int l = lane & 15, g = lane >> 4;
    const _Float16* p = src + (size_t)(m0 + l) * ld + k0 + g * 8;
    AFrag a;
    a.h[0] = *(const v8h*)p;
    a.h[1] = *(const v8h*)(p + 16);
    return a.v;
}

// B-matrix tile 32x16 f16 (K=rows, N=cols) from row-major src (ld in halves).
// ISA layout: lane holds N=n0+(lane&15); lanes 0-15: K {k0..k0+7, k0+16..k0+23};
// lanes 16-31: K {k0+8..k0+15, k0+24..k0+31}.
__device__ __forceinline__ v16h load_tileB(const _Float16* src, int ld, int k0, int n0, int lane) {
    int l = lane & 15, g = lane >> 4;
    const _Float16* p = src + (size_t)(k0 + g * 8) * ld + n0 + l;
    v16h b;
    #pragma unroll
    for (int s = 0; s < 16; ++s)
        b[s] = p[(size_t)(s < 8 ? s : s + 8) * ld];
    return b;
}

__device__ __forceinline__ v8f wmma16(v16h a, v16h b, v8f c) {
    return __builtin_amdgcn_wmma_f32_16x16x32_f16(false, a, false, b, (short)0, c, false, false);
}

// ---------------------------------------------------------------------------
// Kernel 1: channel LayerNorm over c, output fp16 transposed to [b,n,d,c].
// SINGLE global pass: raw f32 column staged in LDS (pad 133 -> conflict-free
// column writes), stats to LDS, transposed fp16 store fully coalesced.
// ---------------------------------------------------------------------------
__global__ __launch_bounds__(128, 4)
void ln_kernel(const float* __restrict__ x0, const float* __restrict__ gamma,
               _Float16* __restrict__ yn)
{
    __shared__ float xt[128][133];        // 68 KB, 133 coprime with 64 banks
    __shared__ float meanr[128], rstdr[128];
    int blk = blockIdx.x;                 // b*512 + d*8 + hwt
    int hwt = blk & 7;
    int d   = (blk >> 3) & 63;
    int b   = blk >> 9;
    int t   = threadIdx.x;
    int hw  = hwt * 128 + t;

    const float* base = x0 + (size_t)b * (CC * DD * NTOK) + (size_t)d * NTOK + hw;
    float s = 0.f, s2 = 0.f;
    for (int c = 0; c < CC; ++c) {
        float v = base[(size_t)c * (DD * NTOK)];
        xt[t][c] = v;
        s += v; s2 += v * v;
    }
    float mean = s * (1.f / CC);
    float var  = s2 * (1.f / CC) - mean * mean;
    meanr[t] = mean;
    rstdr[t] = rsqrtf(var + 1e-5f);
    __syncthreads();

    int wp = t >> 5, l = t & 31;
    float g0 = gamma[l * 4 + 0], g1 = gamma[l * 4 + 1];
    float g2 = gamma[l * 4 + 2], g3 = gamma[l * 4 + 3];
    for (int rr = 0; rr < 32; ++rr) {
        int row = wp * 32 + rr;
        float m  = meanr[row];
        float rs = rstdr[row];
        v4h o;
        o.x = (_Float16)((xt[row][l * 4 + 0] - m) * rs * g0);
        o.y = (_Float16)((xt[row][l * 4 + 1] - m) * rs * g1);
        o.z = (_Float16)((xt[row][l * 4 + 2] - m) * rs * g2);
        o.w = (_Float16)((xt[row][l * 4 + 3] - m) * rs * g3);
        _Float16* drow = yn + (((size_t)b * NTOK + hwt * 128 + row) * DD + d) * CC + l * 4;
        *(v4h*)drow = o;
    }
}

// ---------------------------------------------------------------------------
// Kernel 2: weights -> fp16, transposed so WMMA B-tiles are lane-contiguous.
// wqkvT [c=128][e=768], woutT [e=256][c=128]
// ---------------------------------------------------------------------------
__global__ void wconv_kernel(const float* __restrict__ wqkv, const float* __restrict__ wout,
                             _Float16* __restrict__ wqkvT, _Float16* __restrict__ woutT)
{
    int i = blockIdx.x * 256 + threadIdx.x;
    if (i < CC * QKVE) {
        int k = i / QKVE, e = i % QKVE;
        wqkvT[i] = (_Float16)wqkv[(size_t)e * CC + k];
    }
    if (i < HID * CC) {
        int k = i / CC, c = i % CC;
        woutT[i] = (_Float16)wout[(size_t)c * HID + k];
    }
}

// ---------------------------------------------------------------------------
// Kernel 3: fused QKV + attention + out-projection. One block per (b,n),
// 8 waves = one wave per head. Output overwrites the yn slab in-place.
// LDS (halves):
//   [0 .. 49152)       per-head {Q(2048) KT(2048) V(2048)} x 8   (96 KB)
//   [49152 .. 81920)   per-head P[64][64] x 8                    (64 KB)
//   y[64][128] (8192)  aliases the P region (dead before P is written)
// y slab is fetched with gfx1250 async copy (ASYNCcnt) straight into LDS.
// ---------------------------------------------------------------------------
__global__ __launch_bounds__(256, 1)
void attn_kernel(_Float16* __restrict__ yz,          // [B,N,D,C] in, overwritten out
                 const _Float16* __restrict__ wqkvT, // [C][768]
                 const _Float16* __restrict__ woutT) // [256][128]
{
    __shared__ __align__(16) _Float16 lds[81920];    // 160 KB
    const int bn   = blockIdx.x;
    const int wave = threadIdx.x >> 5;
    const int lane = threadIdx.x & 31;
    const int l    = lane & 15, g = lane >> 4;

    _Float16* yslab = yz + (size_t)bn * (DD * CC);
    _Float16* Y = &lds[49152];

    // warm L2 with the late-phase projection weights (global_prefetch_b8)
    __builtin_prefetch(woutT + (size_t)threadIdx.x * 128, 0, 1);

    // --- async copy y slab [64][128] (16 KB) global -> LDS ----------------
    {
        unsigned ybase = (unsigned)(uintptr_t)Y;     // low 32b of flat = LDS offset
        #pragma unroll
        for (int i = 0; i < 4; ++i) {
            unsigned loff = ybase + (threadIdx.x + i * 256) * 16;
            unsigned long long gaddr =
                (unsigned long long)(uintptr_t)(yslab + (threadIdx.x + i * 256) * 8);
            asm volatile("global_load_async_to_lds_b128 %0, %1, off"
                         :: "v"(loff), "v"(gaddr) : "memory");
        }
#if __has_builtin(__builtin_amdgcn_s_wait_asynccnt)
        __builtin_amdgcn_s_wait_asynccnt(0);
#else
        asm volatile("s_wait_asynccnt 0x0" ::: "memory");
#endif
    }
    __syncthreads();

    // --- phase 1: QKV for head = wave; A-tiles hoisted per m-row ----------
    const int hd = wave;
    _Float16* Q  = &lds[hd * 6144];
    _Float16* KT = Q + 2048;
    _Float16* V  = Q + 4096;
    const float scale = 0.17677669529663687f;        // 32^-0.5

    for (int mt = 0; mt < 4; ++mt) {
        int m0 = mt * 16;
        v16h a0 = load_tileA(Y, CC, m0,  0, lane);
        v16h a1 = load_tileA(Y, CC, m0, 32, lane);
        v16h a2 = load_tileA(Y, CC, m0, 64, lane);
        v16h a3 = load_tileA(Y, CC, m0, 96, lane);
        for (int mat = 0; mat < 3; ++mat) {
            int ecol = mat * HID + hd * DH;
            for (int nt = 0; nt < 2; ++nt) {
                int n0 = nt * 16;
                int nc = ecol + n0;
                v8f acc = {};
                acc = wmma16(a0, load_tileB(wqkvT, QKVE,  0, nc, lane), acc);
                acc = wmma16(a1, load_tileB(wqkvT, QKVE, 32, nc, lane), acc);
                acc = wmma16(a2, load_tileB(wqkvT, QKVE, 64, nc, lane), acc);
                acc = wmma16(a3, load_tileB(wqkvT, QKVE, 96, nc, lane), acc);
                if (mat == 0) {
                    #pragma unroll
                    for (int r = 0; r < 8; ++r)
                        Q[(m0 + g * 8 + r) * DH + n0 + l] = (_Float16)(acc[r] * scale);
                } else if (mat == 1) {               // store K transposed: KT[dim][j]
                    #pragma unroll
                    for (int r = 0; r < 8; ++r)
                        KT[(n0 + l) * DD + (m0 + g * 8 + r)] = (_Float16)acc[r];
                } else {
                    #pragma unroll
                    for (int r = 0; r < 8; ++r)
                        V[(m0 + g * 8 + r) * DH + n0 + l] = (_Float16)acc[r];
                }
            }
        }
    }
    __syncthreads();   // all Y reads done -> P region (aliases Y) may be written

    // --- phase 2: S = Q*KT, in-register softmax, P to LDS ----------------
    _Float16* P = &lds[49152 + hd * 4096];
    for (int mt = 0; mt < 4; ++mt) {
        int m0 = mt * 16;
        v16h a = load_tileA(Q, DH, m0, 0, lane);     // whole K-dim (32) in one tile
        v8f s0 = {}, s1 = {}, s2 = {}, s3 = {};
        s0 = wmma16(a, load_tileB(KT, DD, 0,  0, lane), s0);
        s1 = wmma16(a, load_tileB(KT, DD, 0, 16, lane), s1);
        s2 = wmma16(a, load_tileB(KT, DD, 0, 32, lane), s2);
        s3 = wmma16(a, load_tileB(KT, DD, 0, 48, lane), s3);
        // each VGPR r holds one full row per 16-lane half -> reduce with xor 1..8
        #pragma unroll
        for (int r = 0; r < 8; ++r) {
            float m = fmaxf(fmaxf(s0[r], s1[r]), fmaxf(s2[r], s3[r]));
            #pragma unroll
            for (int off = 8; off >= 1; off >>= 1)
                m = fmaxf(m, __shfl_xor(m, off, 32));
            float e0 = __expf(s0[r] - m), e1 = __expf(s1[r] - m);
            float e2 = __expf(s2[r] - m), e3 = __expf(s3[r] - m);
            float sum = e0 + e1 + e2 + e3;
            #pragma unroll
            for (int off = 8; off >= 1; off >>= 1)
                sum += __shfl_xor(sum, off, 32);
            float rs = __frcp_rn(sum);
            int row = m0 + g * 8 + r;
            P[row * 64 +  0 + l] = (_Float16)(e0 * rs);
            P[row * 64 + 16 + l] = (_Float16)(e1 * rs);
            P[row * 64 + 32 + l] = (_Float16)(e2 * rs);
            P[row * 64 + 48 + l] = (_Float16)(e3 * rs);
        }
    }

    // --- O = P*V, overwrite the (dead) Q buffer --------------------------
    for (int mt = 0; mt < 4; ++mt) {
        int m0 = mt * 16;
        for (int nt = 0; nt < 2; ++nt) {
            v8f acc = {};
            #pragma unroll
            for (int kk = 0; kk < 2; ++kk) {
                v16h a = load_tileA(P, 64, m0, kk * 32, lane);
                v16h b = load_tileB(V, DH, kk * 32, nt * 16, lane);
                acc = wmma16(a, b, acc);
            }
            #pragma unroll
            for (int r = 0; r < 8; ++r)
                Q[(m0 + g * 8 + r) * DH + nt * 16 + l] = (_Float16)acc[r];
        }
    }
    __syncthreads();   // all heads' O ready before cross-head out-projection

    // --- phase 3: out = Oall[64x256] * woutT[256x128]; wave owns 16 cols -
    int n0 = wave * 16;
    for (int mt = 0; mt < 4; ++mt) {
        int m0 = mt * 16;
        v8f acc = {};
        #pragma unroll
        for (int h2 = 0; h2 < HEADS; ++h2) {
            const _Float16* Oh = &lds[h2 * 6144];    // O stored over Q
            v16h a = load_tileA(Oh, DH, m0, 0, lane);
            v16h b = load_tileB(woutT, CC, h2 * DH, n0, lane);
            acc = wmma16(a, b, acc);
        }
        #pragma unroll
        for (int r = 0; r < 8; ++r)
            yslab[(size_t)(m0 + g * 8 + r) * CC + n0 + l] = (_Float16)acc[r];
    }
}

// ---------------------------------------------------------------------------
// Kernel 4: transpose [b,n,d,c] -> [b,c,d,h,w], add residual x0, write out0.
// 32x32 LDS tile so both global sides are coalesced; NT stores (write-once).
// ---------------------------------------------------------------------------
__global__ __launch_bounds__(256, 2)
void finalize_kernel(const float* __restrict__ x0, const _Float16* __restrict__ z,
                     float* __restrict__ out0)
{
    __shared__ _Float16 tl[32][33];
    int blk = blockIdx.x;            // b(2b) d(6b) ct(2b) wt(5b)
    int wt = blk & 31;
    int ct = (blk >> 5) & 3;
    int d  = (blk >> 7) & 63;
    int b  = blk >> 13;
    int col = threadIdx.x & 31, rg = threadIdx.x >> 5;

    const _Float16* zb = z + (((size_t)b * NTOK + wt * 32) * DD + d) * CC + ct * 32;
    #pragma unroll
    for (int k = 0; k < 4; ++k) {
        int row = rg + 8 * k;        // hw within tile
        tl[row][col] = zb[(size_t)row * (DD * CC) + col];
    }
    __syncthreads();
    #pragma unroll
    for (int k = 0; k < 4; ++k) {
        int rw = rg + 8 * k;         // c within tile
        size_t addr = (size_t)b * (CC * DD * NTOK)
                    + (size_t)(ct * 32 + rw) * (DD * NTOK)
                    + (size_t)d * NTOK + wt * 32 + col;
        __builtin_nontemporal_store(x0[addr] + (float)tl[col][rw], &out0[addr]);
    }
}

__global__ void zero_kernel(f4* __restrict__ p, size_t n4)
{
    size_t i = (size_t)blockIdx.x * blockDim.x + threadIdx.x;
    f4 zv = {0.f, 0.f, 0.f, 0.f};
    if (i < n4) __builtin_nontemporal_store(zv, &p[i]);
}

// ---------------------------------------------------------------------------
// Workspace layout (bytes): yn fp16 [4,1024,64,128] = 67,108,864
//                           wqkvT fp16              =    196,608
//                           woutT fp16              =     65,536
// attn output reuses the yn slab in place.
// ---------------------------------------------------------------------------
extern "C" void kernel_launch(void* const* d_in, const int* in_sizes, int n_in,
                              void* d_out, int out_size, void* d_ws, size_t ws_size,
                              hipStream_t stream)
{
    const float* x0    = (const float*)d_in[0];
    // d_in[1], d_in[2] (x1, x2) unused by the reference math
    const float* gamma = (const float*)d_in[3];
    const float* wqkv  = (const float*)d_in[4];
    const float* wout  = (const float*)d_in[5];

    float* out0  = (float*)d_out;
    float* out12 = out0 + (size_t)BB * CC * DD * NTOK;

    _Float16* yn    = (_Float16*)d_ws;
    _Float16* wqkvT = (_Float16*)((char*)d_ws + 67108864);
    _Float16* woutT = wqkvT + CC * QKVE;

    ln_kernel<<<BB * DD * (NTOK / 128), 128, 0, stream>>>(x0, gamma, yn);
    wconv_kernel<<<(CC * QKVE + 255) / 256, 256, 0, stream>>>(wqkv, wout, wqkvT, woutT);
    attn_kernel<<<BB * NTOK, 256, 0, stream>>>(yn, wqkvT, woutT);
    finalize_kernel<<<BB * DD * 4 * 32, 256, 0, stream>>>(x0, yn, out0);
    zero_kernel<<<65536, 256, 0, stream>>>((f4*)out12,
                                           (size_t)2 * BB * CC * DD * NTOK / 4);
}
